// MultiHeadedSelfAttention_858993459369
// MI455X (gfx1250) — matrix operand order repaired and checked by reading, not compiled, not run on hardware
//
#include <hip/hip_runtime.h>
#include <hip/hip_fp16.h>

typedef __attribute__((ext_vector_type(16))) _Float16 v16h;
typedef __attribute__((ext_vector_type(8)))  _Float16 v8h;
typedef __attribute__((ext_vector_type(8)))  float    v8f;
typedef __attribute__((ext_vector_type(4)))  float    f4;
typedef __attribute__((ext_vector_type(4)))  unsigned int u32x4;
typedef __attribute__((ext_vector_type(4)))  int      i32x4;
typedef __attribute__((ext_vector_type(8)))  int      i32x8;

static constexpr int Bb = 4, Ss = 2048, Ee = 512, Hh = 8, Dh = 64;

__device__ __forceinline__ v8f wmma16(v16h a, v16h b, v8f c) {
  // D = A(16x32 f16) * B(32x16 f16) + C(16x16 f32)
  return __builtin_amdgcn_wmma_f32_16x16x32_f16(false, a, false, b, (short)0, c,
                                                false, false);
}

// ---- Tensor Data Mover: 2-D f16 tile (global -> LDS, compacted row-major) ---
// D# packing per CDNA5 ISA:
//  group0: [1:0]=count=1, [63:32]=lds_addr, [120:64]=global_addr, [127:126]=2
//  group1: [17:16]=data_size(1=2B), [79:48]=tensor_dim0, [111:80]=tensor_dim1,
//          [127:112]=tile_dim0, [143:128]=tile_dim1, [207:160]=dim0_stride
__device__ __forceinline__ void tdm_load_tile_f16(unsigned lds_off,
                                                  const void* gptr,
                                                  unsigned d0, unsigned d1,
                                                  unsigned stride0) {
  unsigned long long ga = (unsigned long long)(uintptr_t)gptr;
  u32x4 g0;
  g0[0] = 1u;                                                 // count=1
  g0[1] = lds_off;                                            // LDS byte addr
  g0[2] = (unsigned)ga;                                       // gaddr[31:0]
  g0[3] = (unsigned)((ga >> 32) & 0x1FFFFFFu) | 0x80000000u;  // [56:32]|type=2
  i32x8 g1;
  g1[0] = 0x10000;                                   // data_size = 2 bytes
  g1[1] = (int)((d0 & 0xFFFFu) << 16);               // tensor_dim0 lo
  g1[2] = (int)((d0 >> 16) | ((d1 & 0xFFFFu) << 16));// tensor_dim0 hi|dim1 lo
  g1[3] = (int)((d1 >> 16) | (d0 << 16));            // dim1 hi | tile_dim0
  g1[4] = (int)d1;                                   // tile_dim1 (tile_dim2=0)
  g1[5] = (int)stride0;                              // dim0_stride[31:0]
  g1[6] = 0;
  g1[7] = 0;
  i32x4 z4 = {0, 0, 0, 0};
#if __clang_major__ >= 23
  i32x8 z8 = {0, 0, 0, 0, 0, 0, 0, 0};
  __builtin_amdgcn_tensor_load_to_lds(g0, g1, z4, z4, z8, 0);
#else
  __builtin_amdgcn_tensor_load_to_lds(g0, g1, z4, z4, 0);
#endif
}

// ---- fragment loaders -------------------------------------------------------
// A fragment, 16x32 f16, row-major source (ld elements between rows).
// lane l: row = l&15, K = {kb..kb+7} U {16+kb..16+kb+7}, kb = (l>>4)*8.
__device__ __forceinline__ v16h load_a_f16(const _Float16* __restrict__ src,
                                           int ld, int lane) {
  int r = lane & 15;
  int kb = (lane >> 4) << 3;
  const _Float16* p = src + (size_t)r * ld + kb;
  v8h lo = *(const v8h*)p;
  v8h hi = *(const v8h*)(p + 16);
  v16h a;
#pragma unroll
  for (int i = 0; i < 8; ++i) { a[i] = lo[i]; a[8 + i] = hi[i]; }
  return a;
}

__device__ __forceinline__ v16h load_a_f32(const float* __restrict__ src,
                                           int ld, int lane) {
  int r = lane & 15;
  int kb = (lane >> 4) << 3;
  const float* p = src + (size_t)r * ld + kb;
  f4 x0 = *(const f4*)p;
  f4 x1 = *(const f4*)(p + 4);
  f4 x2 = *(const f4*)(p + 16);
  f4 x3 = *(const f4*)(p + 20);
  v16h a;
#pragma unroll
  for (int i = 0; i < 4; ++i) {
    a[i]      = (_Float16)x0[i];
    a[4 + i]  = (_Float16)x1[i];
    a[8 + i]  = (_Float16)x2[i];
    a[12 + i] = (_Float16)x3[i];
  }
  return a;
}

// B fragment for A*M^T: B[k][n] = M[n][k], M row-major (rows=n, ld elems).
// lane l: n = l&15, K = {kb..kb+7} U {16+kb..16+kb+7}, kb = (l>>4)*8.
__device__ __forceinline__ v16h load_bt_f16(const _Float16* __restrict__ src,
                                            int ld, int lane) {
  int n = lane & 15;
  int kb = (lane >> 4) << 3;
  const _Float16* p = src + (size_t)n * ld + kb;
  v8h lo = *(const v8h*)p;
  v8h hi = *(const v8h*)(p + 16);
  v16h b;
#pragma unroll
  for (int i = 0; i < 8; ++i) { b[i] = lo[i]; b[8 + i] = hi[i]; }
  return b;
}

__device__ __forceinline__ v16h load_bt_f32(const float* __restrict__ src,
                                            int ld, int lane) {
  int n = lane & 15;
  int kb = (lane >> 4) << 3;
  const float* p = src + (size_t)n * ld + kb;
  f4 x0 = *(const f4*)p;
  f4 x1 = *(const f4*)(p + 4);
  f4 x2 = *(const f4*)(p + 16);
  f4 x3 = *(const f4*)(p + 20);
  v16h b;
#pragma unroll
  for (int i = 0; i < 4; ++i) {
    b[i]      = (_Float16)x0[i];
    b[4 + i]  = (_Float16)x1[i];
    b[8 + i]  = (_Float16)x2[i];
    b[12 + i] = (_Float16)x3[i];
  }
  return b;
}

// ---- kernel 1: per-head QKV projection -------------------------------------
// x[B,S,H,Dh] f32 -> Q,K f16 [BH][S][Dh], V f16 transposed [BH][Dh][S].
__global__ __launch_bounds__(256) void qkv_proj_kernel(
    const float* __restrict__ x,
    const float* __restrict__ Wq, const float* __restrict__ bq,
    const float* __restrict__ Wk, const float* __restrict__ bk,
    const float* __restrict__ Wv, const float* __restrict__ bv,
    _Float16* __restrict__ Q, _Float16* __restrict__ K,
    _Float16* __restrict__ Vt) {
  __shared__ __align__(16) _Float16 ldsW[3 * Dh * Dh];
  __shared__ float ldsB[3 * Dh];

  int tid = threadIdx.x;
  int bh = blockIdx.x;
  int b = bh / Hh, h = bh % Hh;

  const float* Ws[3] = {Wq + h * Dh * Dh, Wk + h * Dh * Dh, Wv + h * Dh * Dh};
  const float* bs[3] = {bq + h * Dh, bk + h * Dh, bv + h * Dh};
  for (int idx = tid; idx < 3 * Dh * Dh; idx += 256)
    ldsW[idx] = (_Float16)Ws[idx >> 12][idx & 4095];
  if (tid < 3 * Dh) ldsB[tid] = bs[tid >> 6][tid & 63];
  __syncthreads();

  int lane = tid & 31, wave = tid >> 5;
  int n = lane & 15, g = lane >> 4;
  int s0 = (blockIdx.y * 8 + wave) * 16;

  const float* xp = x + ((size_t)(b * Ss + s0)) * Ee + h * Dh;
  v16h a0 = load_a_f32(xp, Ee, lane);        // X[:, 0:32]
  v16h a1 = load_a_f32(xp + 32, Ee, lane);   // X[:, 32:64]

  _Float16* outs[2] = {Q, K};
#pragma unroll
  for (int m = 0; m < 3; ++m) {
#pragma unroll
    for (int t = 0; t < 4; ++t) {
      float bias = ldsB[m * Dh + t * 16 + n];
      v8f c;
#pragma unroll
      for (int v = 0; v < 8; ++v) c[v] = bias;
      // Y = X * W^T : B[k][n] = W[n][k] -> W rows load contiguously
      c = wmma16(a0, load_bt_f16(ldsW + m * 4096 + t * 16 * Dh, Dh, lane), c);
      c = wmma16(a1, load_bt_f16(ldsW + m * 4096 + t * 16 * Dh + 32, Dh, lane), c);
      if (m < 2) {  // Q,K row-major [S][Dh]
        _Float16* dst =
            outs[m] + ((size_t)bh * Ss + s0 + 8 * g) * Dh + t * 16 + n;
#pragma unroll
        for (int v = 0; v < 8; ++v) dst[(size_t)v * Dh] = (_Float16)c[v];
      } else {      // V transposed [Dh][S]; 8 consecutive s per lane -> 16B store
        _Float16* dst = Vt + ((size_t)bh * Dh + t * 16 + n) * Ss + s0 + 8 * g;
        v8h st;
#pragma unroll
        for (int v = 0; v < 8; ++v) st[v] = (_Float16)c[v];
        *(v8h*)dst = st;
      }
    }
  }
}

// ---- kernel 2: flash attention with TDM double-buffered K/V staging --------
// Block = 4 waves sharing one (b,h); each wave owns a 16-query tile.
// Leader wave DMAs the 32-key K tile (32x64) and V^T tile (64x32) into LDS via
// tensor_load_to_lds (TENSORcnt), double-buffered; 4x less HBM traffic than
// per-wave loads and DMA overlaps WMMA+softmax.
__global__ __launch_bounds__(128) void attn_kernel(
    const _Float16* __restrict__ Q, const _Float16* __restrict__ K,
    const _Float16* __restrict__ Vt, _Float16* __restrict__ O) {
  __shared__ __align__(16) _Float16 ldsK[2][32 * Dh];   // [buf][key][dh]
  __shared__ __align__(16) _Float16 ldsV[2][Dh * 32];   // [buf][dh][key]
  __shared__ __align__(16) _Float16 ldsP[4][16 * 32];   // per-wave P staging

  int tid = threadIdx.x, lane = tid & 31, wave = tid >> 5;
  int n = lane & 15, g = lane >> 4;
  int bh = blockIdx.x;
  int b = bh / Hh, h = bh % Hh;
  int q0 = (blockIdx.y * 4 + wave) * 16;

  const _Float16* Qb = Q + ((size_t)bh * Ss + q0) * Dh;
  v16h aq0 = load_a_f16(Qb, Dh, lane);
  v16h aq1 = load_a_f16(Qb + 32, Dh, lane);

  float mi[8], li[8];
#pragma unroll
  for (int v = 0; v < 8; ++v) { mi[v] = -1e30f; li[v] = 0.0f; }
  v8f o0 = {}, o1 = {}, o2 = {}, o3 = {};

  _Float16* pl = &ldsP[wave][0];
  const float scale = 0.125f;  // 1/sqrt(64)

  const _Float16* Kg = K + (size_t)bh * Ss * Dh;    // rows of 64, stride 64
  const _Float16* Vg = Vt + (size_t)bh * Dh * Ss;   // rows of S, stride S
  unsigned ldsKoff[2] = {(unsigned)(uintptr_t)&ldsK[0][0],
                         (unsigned)(uintptr_t)&ldsK[1][0]};
  unsigned ldsVoff[2] = {(unsigned)(uintptr_t)&ldsV[0][0],
                         (unsigned)(uintptr_t)&ldsV[1][0]};
  bool leader = (wave == 0);

  if (leader) {  // prefetch chunk 0 into buffer 0
    tdm_load_tile_f16(ldsKoff[0], Kg, Dh, 32, Dh);      // 32 keys x 64 dh
    tdm_load_tile_f16(ldsVoff[0], Vg, 32, Dh, Ss);      // 64 dh x 32 keys
  }

  const int NC = Ss / 32;
  for (int i = 0; i < NC; ++i) {
    int buf = i & 1;
    if (leader) {
      if (i + 1 < NC) {
        int kc = (i + 1) * 32;
        tdm_load_tile_f16(ldsKoff[buf ^ 1], Kg + (size_t)kc * Dh, Dh, 32, Dh);
        tdm_load_tile_f16(ldsVoff[buf ^ 1], Vg + kc, 32, Dh, Ss);
        __builtin_amdgcn_s_wait_tensorcnt(2);  // chunk i's 2 in-order ops done
      } else {
        __builtin_amdgcn_s_wait_tensorcnt(0);
      }
    }
    __syncthreads();  // publish LDS tiles to all waves

    const _Float16* Kb = &ldsK[buf][0];
    const _Float16* Vb = &ldsV[buf][0];

    // S = Q * K^T for 32 keys (two 16x16 C tiles)
    v8f s0 = {}, s1 = {};
    s0 = wmma16(aq0, load_bt_f16(Kb, Dh, lane), s0);
    s0 = wmma16(aq1, load_bt_f16(Kb + 32, Dh, lane), s0);
    s1 = wmma16(aq0, load_bt_f16(Kb + 16 * Dh, Dh, lane), s1);
    s1 = wmma16(aq1, load_bt_f16(Kb + 16 * Dh + 32, Dh, lane), s1);

    // online softmax; row v+8g lives across the 16 lanes of this half-wave
#pragma unroll
    for (int v = 0; v < 8; ++v) {
      float e0 = s0[v] * scale, e1 = s1[v] * scale;
      float mx = fmaxf(e0, e1);
      mx = fmaxf(mx, __shfl_xor(mx, 1, 32));
      mx = fmaxf(mx, __shfl_xor(mx, 2, 32));
      mx = fmaxf(mx, __shfl_xor(mx, 4, 32));
      mx = fmaxf(mx, __shfl_xor(mx, 8, 32));
      float mnew = fmaxf(mi[v], mx);
      float p0 = __expf(e0 - mnew), p1 = __expf(e1 - mnew);
      float rs = p0 + p1;
      rs += __shfl_xor(rs, 1, 32);
      rs += __shfl_xor(rs, 2, 32);
      rs += __shfl_xor(rs, 4, 32);
      rs += __shfl_xor(rs, 8, 32);
      float corr = __expf(mi[v] - mnew);
      li[v] = li[v] * corr + rs;
      mi[v] = mnew;
      o0[v] *= corr; o1[v] *= corr; o2[v] *= corr; o3[v] *= corr;
      // stage P (C layout -> LDS row-major 16x32)
      pl[(v + 8 * g) * 32 + n] = (_Float16)p0;
      pl[(v + 8 * g) * 32 + 16 + n] = (_Float16)p1;
    }
    // reload P as A fragment (same-wave DS ops are in order)
    v16h ap = load_a_f16(pl, 32, lane);

    // O += P * V ; V^T tile rows contiguous in-key -> vectorized B loads
    o0 = wmma16(ap, load_bt_f16(Vb + 0 * 16 * 32, 32, lane), o0);
    o1 = wmma16(ap, load_bt_f16(Vb + 1 * 16 * 32, 32, lane), o1);
    o2 = wmma16(ap, load_bt_f16(Vb + 2 * 16 * 32, 32, lane), o2);
    o3 = wmma16(ap, load_bt_f16(Vb + 3 * 16 * 32, 32, lane), o3);

    __syncthreads();  // all waves done reading before buffer is re-filled
  }

  // epilogue: O/l -> attn buffer laid out [B,S,E] (E = H*Dh) in f16
  _Float16* dst = O + ((size_t)(b * Ss + q0 + 8 * g)) * Ee + h * Dh + n;
#pragma unroll
  for (int v = 0; v < 8; ++v) {
    float inv = 1.0f / li[v];
    size_t ro = (size_t)v * Ee;
    dst[ro + 0]  = (_Float16)(o0[v] * inv);
    dst[ro + 16] = (_Float16)(o1[v] * inv);
    dst[ro + 32] = (_Float16)(o2[v] * inv);
    dst[ro + 48] = (_Float16)(o3[v] * inv);
  }
}

// ---- kernel 3: output projection  out = A * Wo^T + bo ----------------------
__global__ __launch_bounds__(256) void out_proj_kernel(
    const _Float16* __restrict__ A, const float* __restrict__ Wo,
    const float* __restrict__ bo, float* __restrict__ out) {
  int tid = threadIdx.x, lane = tid & 31, wave = tid >> 5;
  int n = lane & 15, g = lane >> 4;
  int r0 = blockIdx.x * 16;   // token tile
  int n0 = wave * 64;         // output-column quad (8 waves cover E=512)

  v8f c[4];
#pragma unroll
  for (int t = 0; t < 4; ++t) {
    float bias = bo[n0 + t * 16 + n];
#pragma unroll
    for (int v = 0; v < 8; ++v) c[t][v] = bias;
  }

  const _Float16* Ab = A + (size_t)r0 * Ee;
  for (int kc = 0; kc < Ee; kc += 32) {
    v16h a = load_a_f16(Ab + kc, Ee, lane);
#pragma unroll
    for (int t = 0; t < 4; ++t)
      c[t] = wmma16(a, load_bt_f32(Wo + (size_t)(n0 + t * 16) * Ee + kc, Ee, lane),
                    c[t]);
  }

  float* dst = out + (size_t)(r0 + 8 * g) * Ee + n0 + n;
#pragma unroll
  for (int t = 0; t < 4; ++t)
#pragma unroll
    for (int v = 0; v < 8; ++v) dst[(size_t)v * Ee + t * 16] = c[t][v];
}

// ---- launcher ---------------------------------------------------------------
extern "C" void kernel_launch(void* const* d_in, const int* in_sizes, int n_in,
                              void* d_out, int out_size, void* d_ws,
                              size_t ws_size, hipStream_t stream) {
  const float* x  = (const float*)d_in[0];
  const float* Wq = (const float*)d_in[1];
  const float* bq = (const float*)d_in[2];
  const float* Wk = (const float*)d_in[3];
  const float* bk = (const float*)d_in[4];
  const float* Wv = (const float*)d_in[5];
  const float* bv = (const float*)d_in[6];
  const float* Wo = (const float*)d_in[7];
  const float* bo = (const float*)d_in[8];
  float* out = (float*)d_out;

  size_t qkvElems = (size_t)Bb * Hh * Ss * Dh;  // 4 Mi halves each
  _Float16* Q  = (_Float16*)d_ws;
  _Float16* K  = Q + qkvElems;
  _Float16* Vt = K + qkvElems;
  _Float16* At = Vt + qkvElems;  // attention output, [B,S,E] f16

  qkv_proj_kernel<<<dim3(Bb * Hh, Ss / (16 * 8)), 256, 0, stream>>>(
      x, Wq, bq, Wk, bk, Wv, bv, Q, K, Vt);
  attn_kernel<<<dim3(Bb * Hh, Ss / (16 * 4)), 128, 0, stream>>>(Q, K, Vt, At);
  out_proj_kernel<<<dim3((Bb * Ss) / 16), 256, 0, stream>>>(At, Wo, bo, out);
}
